// NystromAttention_36919538876651
// MI455X (gfx1250) — compile-verified, matrix operand order checked by reading, not run on hardware
//
#include <hip/hip_runtime.h>
#include <hip/hip_bf16.h>

// ---------------------------------------------------------------------------
// Nystrom attention (b=2,h=8,n=8192,d=64, m=256 landmarks) for gfx1250.
// All matmuls via v_wmma_f32_16x16x32_bf16 (bf16 A/B, f32 accumulate).
// ---------------------------------------------------------------------------

typedef __bf16 bf16_t;
typedef bf16_t bf16x8  __attribute__((ext_vector_type(8)));
typedef bf16_t bf16x16 __attribute__((ext_vector_type(16)));
typedef float  floatx8 __attribute__((ext_vector_type(8)));

#define BHh 16      // b*h
#define SEQ 8192    // n
#define DIMd 64     // d
#define LMm 256     // landmarks
#define GRP 32      // n / m
#define NEG_INF (-3.4028235e38f)

// Load one 16x32 bf16 WMMA fragment slice for this lane.
// Caller passes p = rowBase + k0 + half*8; elements 0..7 -> K=+0..7,
// elements 8..15 -> K=+16..23 (per CDNA5 16-bit A/B VGPR layout).
__device__ __forceinline__ bf16x16 load_frag16(const bf16_t* p) {
    bf16x8 lo = *reinterpret_cast<const bf16x8*>(p);
    bf16x8 hi = *reinterpret_cast<const bf16x8*>(p + 16);
    return __builtin_shufflevector(lo, hi, 0,1,2,3,4,5,6,7,8,9,10,11,12,13,14,15);
}

// Generic C = A * B^T. A:[M,K] bf16 row-major, Bt:[N,K] bf16 row-major.
// mode 0: C f32 row-major [M,N]; mode 1: C bf16 row-major [M,N];
// mode 2: C bf16 transposed  [N,M].
// Block: 256 threads = 8 waves; block tile 128(M) x 64(N); wave tile 16x64.
// Requires M%128==0, N%64==0, K%32==0 (true for every call site here).
__global__ __launch_bounds__(256) void gemm_nt_bf16(
    const bf16_t* __restrict__ A, const bf16_t* __restrict__ Bt,
    void* __restrict__ C, int M, int N, int K, int mode,
    long sA, long sB, long sC)
{
    const int bh   = blockIdx.z;
    const bf16_t* Ab = A + (long)bh * sA;
    const bf16_t* Bb = Bt + (long)bh * sB;
    const int lane = threadIdx.x & 31;
    const int wave = threadIdx.x >> 5;
    const int half = lane >> 4;
    const int idx  = lane & 15;

    const int rowA  = blockIdx.x * 128 + wave * 16 + idx;
    const int colB0 = blockIdx.y * 64;

    floatx8 acc[4];
#pragma unroll
    for (int j = 0; j < 4; ++j)
#pragma unroll
        for (int e = 0; e < 8; ++e) acc[j][e] = 0.0f;

    const bf16_t* ap = Ab + (long)rowA * K + half * 8;
    const bf16_t* bp[4];
#pragma unroll
    for (int j = 0; j < 4; ++j)
        bp[j] = Bb + (long)(colB0 + j * 16 + idx) * K + half * 8;

    for (int k0 = 0; k0 < K; k0 += 32) {
        bf16x16 af = load_frag16(ap + k0);
#pragma unroll
        for (int j = 0; j < 4; ++j) {
            bf16x16 bf = load_frag16(bp[j] + k0);
            acc[j] = __builtin_amdgcn_wmma_f32_16x16x32_bf16(
                false, af, false, bf, (short)0, acc[j], false, false);
        }
    }

    // C/D layout: element r of acc holds row (half*8 + r), col = idx (within tile)
    const int mBase = blockIdx.x * 128 + wave * 16 + half * 8;
    if (mode == 0) {
        float* Cb = (float*)C + (long)bh * sC;
#pragma unroll
        for (int j = 0; j < 4; ++j) {
            const int col = colB0 + j * 16 + idx;
#pragma unroll
            for (int r = 0; r < 8; ++r)
                Cb[(long)(mBase + r) * N + col] = acc[j][r];
        }
    } else if (mode == 2) {
        bf16_t* Cb = (bf16_t*)C + (long)bh * sC;
#pragma unroll
        for (int j = 0; j < 4; ++j) {
            const int col = colB0 + j * 16 + idx;
#pragma unroll
            for (int r = 0; r < 8; ++r)
                Cb[(long)col * M + (mBase + r)] = (bf16_t)acc[j][r];
        }
    } else {
        bf16_t* Cb = (bf16_t*)C + (long)bh * sC;
#pragma unroll
        for (int j = 0; j < 4; ++j) {
            const int col = colB0 + j * 16 + idx;
#pragma unroll
            for (int r = 0; r < 8; ++r)
                Cb[(long)(mBase + r) * N + col] = (bf16_t)acc[j][r];
        }
    }
}

// q,k -> bf16 row-major; v -> bf16 transposed [bh, d, n] for attn3@v.
__global__ __launch_bounds__(256) void convert_inputs(
    const float* __restrict__ q, const float* __restrict__ k,
    const float* __restrict__ v,
    bf16_t* __restrict__ qb, bf16_t* __restrict__ kb, bf16_t* __restrict__ vTb)
{
    long id = (long)blockIdx.x * 256 + threadIdx.x;
    if (id >= (long)BHh * SEQ * DIMd) return;
    qb[id] = (bf16_t)q[id];
    kb[id] = (bf16_t)k[id];
    int dd = (int)(id & 63);
    long rest = id >> 6;
    int i  = (int)(rest & (SEQ - 1));
    int bh = (int)(rest >> 13);
    vTb[((long)bh * DIMd + dd) * SEQ + i] = (bf16_t)v[id];
}

// landmark means over groups of 32 rows -> bf16 [bh, m, d]
__global__ __launch_bounds__(256) void landmarks_kernel(
    const float* __restrict__ q, const float* __restrict__ k,
    bf16_t* __restrict__ qlb, bf16_t* __restrict__ klb)
{
    long id = (long)blockIdx.x * 256 + threadIdx.x;
    if (id >= (long)BHh * LMm * DIMd) return;
    int dd = (int)(id & 63);
    long rest = id >> 6;
    int mIdx = (int)(rest & (LMm - 1));
    int bh   = (int)(rest >> 8);
    const float* qp = q + ((long)bh * SEQ + (long)mIdx * GRP) * DIMd + dd;
    const float* kp = k + ((long)bh * SEQ + (long)mIdx * GRP) * DIMd + dd;
    float sq = 0.f, sk = 0.f;
#pragma unroll
    for (int i = 0; i < GRP; ++i) { sq += qp[(long)i * DIMd]; sk += kp[(long)i * DIMd]; }
    qlb[id] = (bf16_t)(sq * (1.0f / GRP));
    klb[id] = (bf16_t)(sk * (1.0f / GRP));
}

// softmax over rows of length 256: one wave per row, 8 elements per lane.
__global__ __launch_bounds__(256) void softmax_rows_256(
    const float* __restrict__ src, bf16_t* __restrict__ dst, long totalRows)
{
    const int lane = threadIdx.x & 31;
    const int wave = threadIdx.x >> 5;
    long row = (long)blockIdx.x * 8 + wave;
    if (row >= totalRows) return;
    const float* p = src + row * 256 + lane * 8;
    float v[8];
    float mx = NEG_INF;
#pragma unroll
    for (int i = 0; i < 8; ++i) { v[i] = p[i]; mx = fmaxf(mx, v[i]); }
    for (int o = 16; o > 0; o >>= 1) mx = fmaxf(mx, __shfl_xor(mx, o, 32));
    float s = 0.f;
#pragma unroll
    for (int i = 0; i < 8; ++i) { v[i] = __expf(v[i] - mx); s += v[i]; }
    for (int o = 16; o > 0; o >>= 1) s += __shfl_xor(s, o, 32);
    const float inv = 1.0f / s;
    bf16_t* q = dst + row * 256 + lane * 8;
#pragma unroll
    for (int i = 0; i < 8; ++i) q[i] = (bf16_t)(v[i] * inv);
}

// softmax over rows of length 8192: one block (256 thr) per row, 32 elems/thread.
__global__ __launch_bounds__(256) void softmax_rows_8192(
    const float* __restrict__ src, bf16_t* __restrict__ dst)
{
    long row = blockIdx.x;
    const float* p = src + row * SEQ;
    const int t = threadIdx.x;
    const int lane = t & 31, wave = t >> 5;
    float v[32];
    float mx = NEG_INF;
#pragma unroll
    for (int i = 0; i < 32; ++i) { v[i] = p[t + i * 256]; mx = fmaxf(mx, v[i]); }
    for (int o = 16; o > 0; o >>= 1) mx = fmaxf(mx, __shfl_xor(mx, o, 32));
    __shared__ float redm[8];
    __shared__ float reds[8];
    if (lane == 0) redm[wave] = mx;
    __syncthreads();
    float m2 = redm[0];
#pragma unroll
    for (int w = 1; w < 8; ++w) m2 = fmaxf(m2, redm[w]);
    float s = 0.f;
#pragma unroll
    for (int i = 0; i < 32; ++i) { v[i] = __expf(v[i] - m2); s += v[i]; }
    for (int o = 16; o > 0; o >>= 1) s += __shfl_xor(s, o, 32);
    if (lane == 0) reds[wave] = s;
    __syncthreads();
    float s2 = 0.f;
#pragma unroll
    for (int w = 0; w < 8; ++w) s2 += reds[w];
    const float inv = 1.0f / s2;
    bf16_t* q = dst + row * SEQ;
#pragma unroll
    for (int i = 0; i < 32; ++i) q[t + i * 256] = (bf16_t)(v[i] * inv);
}

__global__ void init_scalars(unsigned* __restrict__ s) {
    if (threadIdx.x < 2) s[threadIdx.x] = 0u;
}

// max over all (bh) of column sums (smax[0]) and row sums (smax[1]) of attn2.
// Values are positive -> uint bit pattern preserves float ordering.
__global__ __launch_bounds__(256) void pinv_scale_reduce(
    const bf16_t* __restrict__ x, unsigned* __restrict__ smax)
{
    const int bh = blockIdx.x;
    const int j = threadIdx.x;
    const long base = (long)bh * LMm * LMm;
    float cs = 0.f, rs = 0.f;
    for (int i = 0; i < LMm; ++i) {
        cs += (float)x[base + (long)i * LMm + j];   // column-j sum (axis=-2)
        rs += (float)x[base + (long)j * LMm + i];   // row-j sum    (axis=-1)
    }
    atomicMax(&smax[0], __float_as_uint(cs));
    atomicMax(&smax[1], __float_as_uint(rs));
}

// z0 = x^T / (maxcol*maxrow); also store zT = z^T.
__global__ __launch_bounds__(256) void z_init(
    const bf16_t* __restrict__ x, const unsigned* __restrict__ smax,
    bf16_t* __restrict__ z, bf16_t* __restrict__ zT)
{
    long id = (long)blockIdx.x * 256 + threadIdx.x;
    if (id >= (long)BHh * LMm * LMm) return;
    int j = (int)(id & (LMm - 1));
    long r = id >> 8;
    int i  = (int)(r & (LMm - 1));
    int bh = (int)(r >> 8);
    const float inv = 1.0f / (__uint_as_float(smax[0]) * __uint_as_float(smax[1]));
    const long base = (long)bh * LMm * LMm;
    z[id]  = (bf16_t)((float)x[base + (long)j * LMm + i] * inv);
    zT[id] = (bf16_t)((float)x[id] * inv);
}

// From f32 M: optionally bf16 row-major copy; always out_t = (alpha*I - M)^T bf16.
__global__ __launch_bounds__(256) void pinv_elem(
    const float* __restrict__ Mf, bf16_t* __restrict__ out_row,
    bf16_t* __restrict__ out_t, float alpha)
{
    long id = (long)blockIdx.x * 256 + threadIdx.x;
    if (id >= (long)BHh * LMm * LMm) return;
    int j = (int)(id & (LMm - 1));
    long r = id >> 8;
    int i  = (int)(r & (LMm - 1));
    int bh = (int)(r >> 8);
    float m = Mf[id];
    if (out_row) out_row[id] = (bf16_t)m;
    float tv = ((i == j) ? alpha : 0.0f) - m;
    out_t[(long)bh * LMm * LMm + (long)j * LMm + i] = (bf16_t)tv;
}

// z = 0.25 * Zf ; also store zT.
__global__ __launch_bounds__(256) void z_convert(
    const float* __restrict__ Zf, bf16_t* __restrict__ z, bf16_t* __restrict__ zT)
{
    long id = (long)blockIdx.x * 256 + threadIdx.x;
    if (id >= (long)BHh * LMm * LMm) return;
    int j = (int)(id & (LMm - 1));
    long r = id >> 8;
    int i  = (int)(r & (LMm - 1));
    int bh = (int)(r >> 8);
    float zn = 0.25f * Zf[id];
    z[id] = (bf16_t)zn;
    zT[(long)bh * LMm * LMm + (long)j * LMm + i] = (bf16_t)zn;
}

// out += depthwise conv over sequence dim of v (kernel 33, pad 16, correlation).
__global__ __launch_bounds__(256) void conv_residual_add(
    float* __restrict__ out, const float* __restrict__ v,
    const float* __restrict__ w, long total)
{
    long id = (long)blockIdx.x * 256 + threadIdx.x;
    if (id >= total) return;
    int dd = (int)(id & 63);
    long rest = id >> 6;
    int i  = (int)(rest & (SEQ - 1));
    int bh = (int)(rest >> 13);
    int h = bh & 7;
    const float* vb = v + (long)bh * SEQ * DIMd + dd;
    float acc = out[id];
#pragma unroll
    for (int t = 0; t < 33; ++t) {
        int ii = i + t - 16;
        if (ii >= 0 && ii < SEQ) acc += w[h * 33 + t] * vb[(long)ii * DIMd];
    }
    out[id] = acc;
}

extern "C" void kernel_launch(void* const* d_in, const int* in_sizes, int n_in,
                              void* d_out, int out_size, void* d_ws, size_t ws_size,
                              hipStream_t stream)
{
    (void)in_sizes; (void)out_size;
    if (n_in < 4) return;
    const float* q = (const float*)d_in[0];
    const float* k = (const float*)d_in[1];
    const float* v = (const float*)d_in[2];
    const float* w = (const float*)d_in[3];
    float* out = (float*)d_out;

    // bump allocator over workspace (256B aligned)
    size_t off = 0;
    auto alloc = [&](size_t bytes) -> void* {
        void* p = (char*)d_ws + off;
        off += (bytes + 255) & ~(size_t)255;
        return p;
    };
    bf16_t* qb     = (bf16_t*)alloc((size_t)BHh * SEQ * DIMd * 2);
    bf16_t* kb     = (bf16_t*)alloc((size_t)BHh * SEQ * DIMd * 2);
    bf16_t* vTb    = (bf16_t*)alloc((size_t)BHh * DIMd * SEQ * 2);
    bf16_t* qlb    = (bf16_t*)alloc((size_t)BHh * LMm * DIMd * 2);
    bf16_t* klb    = (bf16_t*)alloc((size_t)BHh * LMm * DIMd * 2);
    bf16_t* attn1b = (bf16_t*)alloc((size_t)BHh * SEQ * LMm * 2);
    bf16_t* attn3b = (bf16_t*)alloc((size_t)BHh * LMm * SEQ * 2);
    bf16_t* xb     = (bf16_t*)alloc((size_t)BHh * LMm * LMm * 2);
    bf16_t* zb     = (bf16_t*)alloc((size_t)BHh * LMm * LMm * 2);
    bf16_t* zTb    = (bf16_t*)alloc((size_t)BHh * LMm * LMm * 2);
    bf16_t* xzb    = (bf16_t*)alloc((size_t)BHh * LMm * LMm * 2);
    bf16_t* tmpT   = (bf16_t*)alloc((size_t)BHh * LMm * LMm * 2);
    bf16_t* t1t    = (bf16_t*)alloc((size_t)BHh * DIMd * LMm * 2);
    bf16_t* t2t    = (bf16_t*)alloc((size_t)BHh * DIMd * LMm * 2);
    unsigned* scal = (unsigned*)alloc(256);
    float* simf    = (float*)alloc((size_t)BHh * SEQ * LMm * 4); // f32 logits, reused
    float* pinvf   = simf;                                       // aliased (sequential use)
    if (off > ws_size) return; // workspace too small: bail safely

    const dim3 blk(256, 1, 1);
    const long ZEL = (long)BHh * LMm * LMm;   // 1,048,576
    const long TEL = (long)BHh * SEQ * DIMd;  // 8,388,608

    init_scalars<<<1, 32, 0, stream>>>(scal);
    convert_inputs<<<(unsigned)(TEL / 256), blk, 0, stream>>>(q, k, v, qb, kb, vTb);
    landmarks_kernel<<<(unsigned)((BHh * LMm * DIMd) / 256), blk, 0, stream>>>(q, k, qlb, klb);

    // sim2 = q_land @ k_land^T -> softmax -> x (attn2, bf16)
    gemm_nt_bf16<<<dim3(LMm / 128, LMm / 64, BHh), blk, 0, stream>>>(
        qlb, klb, simf, LMm, LMm, DIMd, 0,
        (long)LMm * DIMd, (long)LMm * DIMd, (long)LMm * LMm);
    softmax_rows_256<<<(unsigned)((BHh * LMm) / 8), blk, 0, stream>>>(simf, xb, (long)BHh * LMm);
    pinv_scale_reduce<<<BHh, LMm, 0, stream>>>(xb, scal);
    z_init<<<(unsigned)(ZEL / 256), blk, 0, stream>>>(xb, scal, zb, zTb);

    // sim1 = q @ k_land^T -> softmax -> attn1
    gemm_nt_bf16<<<dim3(SEQ / 128, LMm / 64, BHh), blk, 0, stream>>>(
        qb, klb, simf, SEQ, LMm, DIMd, 0,
        (long)SEQ * DIMd, (long)LMm * DIMd, (long)SEQ * LMm);
    softmax_rows_256<<<(unsigned)((BHh * SEQ) / 8), blk, 0, stream>>>(simf, attn1b, (long)BHh * SEQ);

    // sim3 = q_land @ k^T -> softmax -> attn3
    gemm_nt_bf16<<<dim3(LMm / 128, SEQ / 64, BHh), blk, 0, stream>>>(
        qlb, kb, simf, LMm, SEQ, DIMd, 0,
        (long)LMm * DIMd, (long)SEQ * DIMd, (long)LMm * SEQ);
    softmax_rows_8192<<<(unsigned)(BHh * LMm), blk, 0, stream>>>(simf, attn3b);

    // t1^T = (attn3 @ v)^T  (bf16, transposed store)
    gemm_nt_bf16<<<dim3(LMm / 128, DIMd / 64, BHh), blk, 0, stream>>>(
        attn3b, vTb, t1t, LMm, DIMd, SEQ, 2,
        (long)LMm * SEQ, (long)DIMd * SEQ, (long)DIMd * LMm);

    // Moore-Penrose iterations: z = 0.25 z (13I - xz (15I - xz (7I - xz)))
    for (int it = 0; it < 6; ++it) {
        gemm_nt_bf16<<<dim3(LMm / 128, LMm / 64, BHh), blk, 0, stream>>>(
            xb, zTb, pinvf, LMm, LMm, LMm, 0, ZEL / BHh, ZEL / BHh, ZEL / BHh);         // xz
        pinv_elem<<<(unsigned)(ZEL / 256), blk, 0, stream>>>(pinvf, xzb, tmpT, 7.0f);   // P^T
        gemm_nt_bf16<<<dim3(LMm / 128, LMm / 64, BHh), blk, 0, stream>>>(
            xzb, tmpT, pinvf, LMm, LMm, LMm, 0, ZEL / BHh, ZEL / BHh, ZEL / BHh);       // xz@P
        pinv_elem<<<(unsigned)(ZEL / 256), blk, 0, stream>>>(pinvf, (bf16_t*)nullptr, tmpT, 15.0f); // R^T
        gemm_nt_bf16<<<dim3(LMm / 128, LMm / 64, BHh), blk, 0, stream>>>(
            xzb, tmpT, pinvf, LMm, LMm, LMm, 0, ZEL / BHh, ZEL / BHh, ZEL / BHh);       // xz@R
        pinv_elem<<<(unsigned)(ZEL / 256), blk, 0, stream>>>(pinvf, (bf16_t*)nullptr, tmpT, 13.0f); // T^T
        gemm_nt_bf16<<<dim3(LMm / 128, LMm / 64, BHh), blk, 0, stream>>>(
            zb, tmpT, pinvf, LMm, LMm, LMm, 0, ZEL / BHh, ZEL / BHh, ZEL / BHh);        // z@T
        z_convert<<<(unsigned)(ZEL / 256), blk, 0, stream>>>(pinvf, zb, zTb);
    }

    // t2^T = (z @ t1)^T
    gemm_nt_bf16<<<dim3(LMm / 128, DIMd / 64, BHh), blk, 0, stream>>>(
        zb, t1t, t2t, LMm, DIMd, LMm, 2,
        ZEL / BHh, (long)DIMd * LMm, (long)DIMd * LMm);

    // out = attn1 @ t2 (f32, straight into d_out)
    gemm_nt_bf16<<<dim3(SEQ / 128, DIMd / 64, BHh), blk, 0, stream>>>(
        attn1b, t2t, out, SEQ, DIMd, LMm, 0,
        (long)SEQ * LMm, (long)DIMd * LMm, (long)SEQ * DIMd);

    // out += depthwise conv residual
    conv_residual_add<<<(unsigned)(TEL / 256), blk, 0, stream>>>(out, v, w, TEL);
}
